// Decoder_88871463288984
// MI455X (gfx1250) — matrix operand order repaired and checked by reading, not compiled, not run on hardware
//
#include <hip/hip_runtime.h>
#include <cstdint>

// ---------------- problem constants ----------------
#define LNUM 6
#define HNUM 8
#define MDIM 1024
#define DHEAD 128
#define FDIM 4096
#define BDIM 4
#define QDIM 1024
#define TDIM 1024

typedef uint16_t u16;
typedef uint32_t u32;
typedef uint64_t u64;

typedef float  fvec8  __attribute__((ext_vector_type(8)));
typedef __bf16 bvec16 __attribute__((ext_vector_type(16)));

union BFrag { u32 u[8]; bvec16 v; };

// ---------------- bf16 helpers (round-to-nearest-even) ----------------
__device__ inline u16 f2bf(float f) {
  union { float f; u32 u; } c; c.f = f;
  u32 r = c.u + 0x7FFFu + ((c.u >> 16) & 1u);
  return (u16)(r >> 16);
}

// ---------------- CDNA5 async copy global->LDS (ASYNCcnt tracked) ----------------
// ldsOff: LDS byte address (low 32 bits of flat shared pointer); ga: global byte address.
__device__ inline void async_copy_b32(u32 ldsOff, u64 ga) {
  asm volatile("global_load_async_to_lds_b32 %0, %1, off"
               :: "v"(ldsOff), "v"(ga) : "memory");
}
__device__ inline void wait_async0() {
  asm volatile("s_wait_asynccnt 0x0" ::: "memory");
}

// ---------------- WMMA fragment loaders (CDNA5 wave32 layouts) ----------------
// A-matrix 16x32 bf16: lanes 0-15 -> rows, VGPR i holds K pairs:
//   lanes 0-15 : K = {0,2,4,6} and {16..22};  lanes 16-31: K = {8..14} and {24..30}
__device__ inline void load_a16x32(const u16* base, int stride, int row0, int k0, BFrag& f) {
  const int lane = threadIdx.x & 31;
  const u16* p = base + (size_t)(row0 + (lane & 15)) * stride + k0 + ((lane >> 4) << 3);
#pragma unroll
  for (int i = 0; i < 4; ++i) {
    f.u[i]     = *(const u32*)(p + 2 * i);
    f.u[i + 4] = *(const u32*)(p + 2 * i + 16);
  }
}

// B-matrix 32x16 bf16: lanes 0-15 -> cols N, K order matches A (half h: K = 8h..8h+7, 16+8h..16+8h+7)
__device__ inline void load_b32x16(const u16* base, int stride, int k0, int n0, BFrag& f) {
  const int lane = threadIdx.x & 31;
  const u16* p = base + (size_t)(k0 + ((lane >> 4) << 3)) * stride + n0 + (lane & 15);
#pragma unroll
  for (int i = 0; i < 4; ++i) {
    f.u[i]     = (u32)p[(size_t)(2 * i) * stride] | ((u32)p[(size_t)(2 * i + 1) * stride] << 16);
    f.u[i + 4] = (u32)p[(size_t)(2 * i + 16) * stride] | ((u32)p[(size_t)(2 * i + 17) * stride] << 16);
  }
}

__device__ inline fvec8 wmma_bf16(const BFrag& a, const BFrag& b, fvec8 c) {
  return __builtin_amdgcn_wmma_f32_16x16x32_bf16(false, a.v, false, b.v,
                                                 (short)0, c, false, false);
}

// ---------------- weight cast / repack kernels ----------------
__global__ void cast_flat_kernel(const float* __restrict__ in, u16* __restrict__ out, size_t n) {
  size_t i = (size_t)blockIdx.x * blockDim.x + threadIdx.x;
  const size_t stride = (size_t)gridDim.x * blockDim.x;
  for (; i < n; i += stride) out[i] = f2bf(in[i]);
}

// (L,H,M,128) f32  ->  (L, M, H*128) bf16
__global__ void cast_qkvw_kernel(const float* __restrict__ in, u16* __restrict__ out) {
  const size_t n = (size_t)LNUM * HNUM * MDIM * DHEAD;
  size_t i = (size_t)blockIdx.x * blockDim.x + threadIdx.x;
  const size_t stride = (size_t)gridDim.x * blockDim.x;
  for (; i < n; i += stride) {
    size_t d = i & 127;
    size_t m = (i >> 7) & (MDIM - 1);
    size_t h = (i >> 17) & (HNUM - 1);
    size_t l = i >> 20;
    out[(l * MDIM + m) * (HNUM * DHEAD) + h * DHEAD + d] = f2bf(in[i]);
  }
}

// ---------------- LayerNorm (f32 in -> bf16 out), one row per block ----------------
__global__ __launch_bounds__(256) void ln_kernel(const float* __restrict__ x,
                                                 const float* __restrict__ g,
                                                 const float* __restrict__ bta,
                                                 u16* __restrict__ out) {
  const int row = blockIdx.x;
  const int tid = threadIdx.x;
  const float* px = x + (size_t)row * MDIM;
  float v[4];
  float s = 0.f, ss = 0.f;
#pragma unroll
  for (int j = 0; j < 4; ++j) {
    v[j] = px[tid + 256 * j];
    s += v[j]; ss += v[j] * v[j];
  }
  __shared__ float red[16];
#pragma unroll
  for (int d = 1; d < 32; d <<= 1) { s += __shfl_xor(s, d, 32); ss += __shfl_xor(ss, d, 32); }
  const int wave = tid >> 5, lane = tid & 31;
  if (lane == 0) { red[wave] = s; red[8 + wave] = ss; }
  __syncthreads();
  if (wave == 0) {
    float a = (lane < 8) ? red[lane] : 0.f;
    float b = (lane < 8) ? red[8 + lane] : 0.f;
#pragma unroll
    for (int d = 1; d < 8; d <<= 1) { a += __shfl_xor(a, d, 32); b += __shfl_xor(b, d, 32); }
    if (lane == 0) { red[0] = a; red[1] = b; }
  }
  __syncthreads();
  const float mu = red[0] * (1.0f / MDIM);
  const float var = red[1] * (1.0f / MDIM) - mu * mu;
  const float rstd = rsqrtf(var + 1e-5f);
#pragma unroll
  for (int j = 0; j < 4; ++j) {
    int i = tid + 256 * j;
    out[(size_t)row * MDIM + i] = f2bf((v[j] - mu) * rstd * g[i] + bta[i]);
  }
}

// ---------------- generic bf16 WMMA GEMM: C[rows x N] = A[rows x Kd] * W[Kd x N] ----------------
// Double-buffered LDS stages filled by global_load_async_to_lds_b32 (ASYNCcnt).
// Epilogue: +bias, ReLU, *(1-rowmask), +residual; store f32 (outF) or bf16 (outB).
__global__ __launch_bounds__(256) void gemm_bf16_kernel(
    const u16* __restrict__ A, const u16* __restrict__ W, int Kd, int N,
    const float* __restrict__ bias, const float* __restrict__ rowmask,
    const float* __restrict__ residual, float* __restrict__ outF,
    u16* __restrict__ outB, int relu) {
  __shared__ u16 As[2][128 * 34];   // stride 34 halfwords (68B = 17 banks)
  __shared__ u16 Bs[2][32 * 130];   // stride 130 halfwords (260B = 65 banks)
  const int tid = threadIdx.x;
  const int wave = tid >> 5, lane = tid & 31;
  const int wr = wave >> 1, wc = wave & 1;   // wave computes 32x64 sub-tile
  const int row0 = blockIdx.y * 128, col0 = blockIdx.x * 128;

  // Hoisted per-thread staging addresses.
  // A slots: idx = tid + 256*j -> row = (tid>>4) + 16j, kpair = tid&15 (j-invariant)
  // B slots: idx = tid + 256*j -> krow = (tid>>6) + 4j, npair = tid&63 (j-invariant)
  u64 aG = (u64)(uintptr_t)(A + (size_t)(row0 + (tid >> 4)) * Kd + 2 * (tid & 15));
  u64 bG = (u64)(uintptr_t)(W + (size_t)(tid >> 6) * N + col0 + 2 * (tid & 63));
  const u64 aStepJ = (u64)16 * Kd * 2;     // +16 rows
  const u64 bStepJ = (u64)4 * N * 2;       // +4 k-rows
  const u64 aStepK = 64;                   // +32 k elems
  const u64 bStepK = (u64)32 * N * 2;      // +32 k-rows
  const u32 aL0 = (u32)(uintptr_t)(&As[0][(tid >> 4) * 34 + 2 * (tid & 15)]);
  const u32 bL0 = (u32)(uintptr_t)(&Bs[0][(tid >> 6) * 130 + 2 * (tid & 63)]);
  const u32 aLbuf = 128 * 34 * 2, bLbuf = 32 * 130 * 2;   // buffer byte strides
  const u32 aLj = 16 * 34 * 2,  bLj = 4 * 130 * 2;        // per-j byte strides

  auto stage = [&](int buf, u64 ga, u64 gb) {
    u32 la = aL0 + (u32)buf * aLbuf;
    u32 lb = bL0 + (u32)buf * bLbuf;
#pragma unroll
    for (int j = 0; j < 8; ++j) {
      async_copy_b32(la, ga);
      async_copy_b32(lb, gb);
      ga += aStepJ; la += aLj;
      gb += bStepJ; lb += bLj;
    }
  };

  fvec8 acc[2][4];
#pragma unroll
  for (int i = 0; i < 2; ++i)
#pragma unroll
    for (int j = 0; j < 4; ++j)
      acc[i][j] = (fvec8){0.f, 0.f, 0.f, 0.f, 0.f, 0.f, 0.f, 0.f};

  stage(0, aG, bG);
  int buf = 0;
  for (int k0 = 0; k0 < Kd; k0 += 32) {
    wait_async0();        // this wave's stage(buf) loads have landed in LDS
    __syncthreads();      // every wave's loads landed; prev compute of buf^1 done
    aG += aStepK; bG += bStepK;
    if (k0 + 32 < Kd) stage(buf ^ 1, aG, bG);  // overlap next stage with WMMAs

    const u16* Asb = As[buf];
    const u16* Bsb = Bs[buf];
    BFrag af[2], bfm[4];
    load_a16x32(Asb, 34, wr * 32, 0, af[0]);
    load_a16x32(Asb, 34, wr * 32 + 16, 0, af[1]);
#pragma unroll
    for (int j = 0; j < 4; ++j) load_b32x16(Bsb, 130, 0, wc * 64 + j * 16, bfm[j]);
#pragma unroll
    for (int i = 0; i < 2; ++i)
#pragma unroll
      for (int j = 0; j < 4; ++j)
        acc[i][j] = wmma_bf16(af[i], bfm[j], acc[i][j]);
    buf ^= 1;
  }

  // epilogue; C layout: lanes 0-15 N=c rows M=v, lanes 16-31 N=c rows M=v+8
  const int half = lane >> 4, c = lane & 15;
#pragma unroll
  for (int i = 0; i < 2; ++i) {
#pragma unroll
    for (int j = 0; j < 4; ++j) {
      const int n = col0 + wc * 64 + j * 16 + c;
      const float bv = bias ? bias[n] : 0.0f;
#pragma unroll
      for (int v = 0; v < 8; ++v) {
        const int m = row0 + wr * 32 + i * 16 + v + half * 8;
        float val = acc[i][j][v] + bv;
        if (relu) val = fmaxf(val, 0.0f);
        if (rowmask) val *= (1.0f - rowmask[m]);
        if (residual) val += residual[(size_t)m * N + n];
        if (outF) outF[(size_t)m * N + n] = val;
        else      outB[(size_t)m * N + n] = f2bf(val);
      }
    }
  }
}

// ---------------- flash attention: one (b,h) x 128 q-rows per block ----------------
// Qm/Km/Vm/Om: bf16 [B, rows, H*128]. Causal = analytic q<t mask. Softmax f32 online.
__global__ __launch_bounds__(256) void attn_kernel(
    const u16* __restrict__ Qm, const u16* __restrict__ Km, const u16* __restrict__ Vm,
    u16* __restrict__ Om, int Tlen, int causal) {
  __shared__ u16 kT[128 * 66];      // K transposed [d][t], stride 66 (33 banks)
  __shared__ u16 Vs[64 * 130];      // V row-major [t][d], stride 130
  __shared__ u16 Ps[8][16 * 66];    // per-wave private P scratch (C->A relayout)
  const int tid = threadIdx.x, wave = tid >> 5, lane = tid & 31;
  const int qt = blockIdx.x, bh = blockIdx.y;
  const int b = bh >> 3, h = bh & 7;
  const int q0 = qt * 128;
  const int half = lane >> 4, c = lane & 15;
  const int RS = HNUM * DHEAD;      // 1024 row stride

  // Q fragments: 16 rows per wave, full d=128, resident in VGPRs
  BFrag qf[4];
  {
    const u16* base = Qm + (size_t)b * QDIM * RS + (size_t)h * DHEAD;
#pragma unroll
    for (int s = 0; s < 4; ++s)
      load_a16x32(base, RS, q0 + wave * 16, s * 32, qf[s]);
  }

  fvec8 o[8];
#pragma unroll
  for (int j = 0; j < 8; ++j) o[j] = (fvec8){0.f, 0.f, 0.f, 0.f, 0.f, 0.f, 0.f, 0.f};
  float mrow[8], lrow[8];
#pragma unroll
  for (int v = 0; v < 8; ++v) { mrow[v] = -3.0e38f; lrow[v] = 0.0f; }

  // Hoisted staging addresses: slot idx = tid + 256*j -> t = (tid>>6)+4j, dpair = tid&63
  const u16* kBase = Km + ((size_t)b * Tlen + (tid >> 6)) * RS + h * DHEAD + 2 * (tid & 63);
  u64 vG = (u64)(uintptr_t)(Vm + ((size_t)b * Tlen + (tid >> 6)) * RS + h * DHEAD + 2 * (tid & 63));
  const u32 vL0 = (u32)(uintptr_t)(&Vs[(tid >> 6) * 130 + 2 * (tid & 63)]);
  u16* kTbase = &kT[(tid & 63) * 132 + (tid >> 6)];

  const float scale = 0.0883883476483184f;  // 1/sqrt(128)
  int tend = causal ? (q0 + 128) : Tlen;
  if (tend > Tlen) tend = Tlen;

  for (int t0 = 0; t0 < tend; t0 += 64) {
    __syncthreads();
    // stage K (register transpose -> LDS) and V (async direct to LDS)
    {
      const u16* kg = kBase;
      u64 vg = vG;
      u32 vl = vL0;
      u16* kt = kTbase;
#pragma unroll
      for (int j = 0; j < 16; ++j) {
        u32 wk = *(const u32*)kg;
        kt[0]  = (u16)(wk & 0xFFFFu);
        kt[66] = (u16)(wk >> 16);
        async_copy_b32(vl, vg);
        kg += 4 * RS; kt += 4;
        vg += (u64)4 * RS * 2; vl += 4 * 130 * 2;
      }
    }
    kBase += (size_t)64 * RS;
    vG += (u64)64 * RS * 2;
    wait_async0();
    __syncthreads();

    // S = Q x K^T  (16 q-rows x 64 t-cols per wave)
    fvec8 s[4];
#pragma unroll
    for (int j = 0; j < 4; ++j) s[j] = (fvec8){0.f, 0.f, 0.f, 0.f, 0.f, 0.f, 0.f, 0.f};
#pragma unroll
    for (int ds = 0; ds < 4; ++ds) {
#pragma unroll
      for (int j = 0; j < 4; ++j) {
        BFrag bf;
        load_b32x16(kT, 66, ds * 32, j * 16, bf);
        s[j] = wmma_bf16(qf[ds], bf, s[j]);
      }
    }

    // online softmax (f32), row = VGPR index (+8 for upper lanes)
    float mnew[8];
#pragma unroll
    for (int v = 0; v < 8; ++v) {
      const int qrow = q0 + wave * 16 + v + half * 8;
      float mx = -3.0e38f;
#pragma unroll
      for (int j = 0; j < 4; ++j) {
        const int tc = t0 + j * 16 + c;
        float z = s[j][v] * scale;
        if (causal && qrow < tc) z = -3.0e38f;
        s[j][v] = z;
        mx = fmaxf(mx, z);
      }
#pragma unroll
      for (int d = 1; d < 16; d <<= 1) mx = fmaxf(mx, __shfl_xor(mx, d, 32));
      mnew[v] = fmaxf(mrow[v], mx);
    }
#pragma unroll
    for (int v = 0; v < 8; ++v) {
      float alpha = __expf(mrow[v] - mnew[v]);
      if (mrow[v] <= -1.0e38f) alpha = 0.0f;
      float rs = 0.0f;
      const int r = v + half * 8;
#pragma unroll
      for (int j = 0; j < 4; ++j) {
        float p = (s[j][v] <= -1.0e38f) ? 0.0f : __expf(s[j][v] - mnew[v]);
        rs += p;
        Ps[wave][r * 66 + j * 16 + c] = f2bf(p);  // wave-private: no barrier needed
      }
#pragma unroll
      for (int d = 1; d < 16; d <<= 1) rs += __shfl_xor(rs, d, 32);
      lrow[v] = lrow[v] * alpha + rs;
      mrow[v] = mnew[v];
#pragma unroll
      for (int j = 0; j < 8; ++j) o[j][v] *= alpha;
    }

    // O += P x V
#pragma unroll
    for (int ts = 0; ts < 2; ++ts) {
      BFrag pa;
      load_a16x32(Ps[wave], 66, 0, ts * 32, pa);
#pragma unroll
      for (int j = 0; j < 8; ++j) {
        BFrag bv;
        load_b32x16(Vs, 130, ts * 32, j * 16, bv);
        o[j] = wmma_bf16(pa, bv, o[j]);
      }
    }
  }

  // normalize + store bf16 context [B, Q, H*128]
#pragma unroll
  for (int v = 0; v < 8; ++v) {
    const int q = q0 + wave * 16 + v + half * 8;
    const float inv = (lrow[v] > 0.0f) ? 1.0f / lrow[v] : 0.0f;
#pragma unroll
    for (int j = 0; j < 8; ++j)
      Om[((size_t)b * QDIM + q) * RS + h * DHEAD + j * 16 + c] = f2bf(o[j][v] * inv);
  }
}

// ---------------- launcher ----------------
extern "C" void kernel_launch(void* const* d_in, const int* in_sizes, int n_in,
                              void* d_out, int out_size, void* d_ws, size_t ws_size,
                              hipStream_t stream) {
  (void)in_sizes; (void)n_in; (void)out_size; (void)ws_size;
  const float* enc   = (const float*)d_in[0];
  const float* x     = (const float*)d_in[1];
  const float* qmask = (const float*)d_in[2];
  const float* swq = (const float*)d_in[5];
  const float* swk = (const float*)d_in[6];
  const float* swv = (const float*)d_in[7];
  const float* swo = (const float*)d_in[8];
  const float* cwq = (const float*)d_in[9];
  const float* cwk = (const float*)d_in[10];
  const float* cwv = (const float*)d_in[11];
  const float* cwo = (const float*)d_in[12];
  const float* w1  = (const float*)d_in[13];
  const float* b1  = (const float*)d_in[14];
  const float* w2  = (const float*)d_in[15];
  const float* b2  = (const float*)d_in[16];
  const float* ln1g = (const float*)d_in[17];
  const float* ln1b = (const float*)d_in[18];
  const float* ln2g = (const float*)d_in[19];
  const float* ln2b = (const float*)d_in[20];
  const float* ln3g = (const float*)d_in[21];
  const float* ln3b = (const float*)d_in[22];

  // workspace carve (bump allocator)
  char* p = (char*)d_ws;
  auto alloc = [&](size_t bytes) -> char* {
    char* r = p; p += (bytes + 255) & ~(size_t)255; return r;
  };
  const size_t wAtt = (size_t)LNUM * MDIM * (HNUM * DHEAD) * sizeof(u16); // 12.6 MB
  const size_t wFfn = (size_t)LNUM * MDIM * FDIM * sizeof(u16);           // 50.3 MB
  const size_t actB = (size_t)BDIM * QDIM * MDIM * sizeof(u16);           // 8.4 MB
  const size_t actF = (size_t)BDIM * QDIM * MDIM * sizeof(float);         // 16.8 MB

  u16* Wq_s = (u16*)alloc(wAtt); u16* Wk_s = (u16*)alloc(wAtt); u16* Wv_s = (u16*)alloc(wAtt);
  u16* Wo_s = (u16*)alloc(wAtt);
  u16* Wq_c = (u16*)alloc(wAtt); u16* Wk_c = (u16*)alloc(wAtt); u16* Wv_c = (u16*)alloc(wAtt);
  u16* Wo_c = (u16*)alloc(wAtt);
  u16* W1b  = (u16*)alloc(wFfn); u16* W2b  = (u16*)alloc(wFfn);
  u16* encB = (u16*)alloc(actB);
  u16* nB   = (u16*)alloc(actB);
  u16* qB   = (u16*)alloc(actB);
  u16* kB   = (u16*)alloc(actB);
  u16* vB   = (u16*)alloc(actB);
  u16* preB = (u16*)alloc(actB);
  u16* sB   = (u16*)alloc((size_t)BDIM * QDIM * FDIM * sizeof(u16));
  float* h  = (float*)alloc(actF);
  float* h1 = (float*)alloc(actF);
  float* h2 = (float*)alloc(actF);

  const dim3 blk(256);
  const dim3 castGrid(2048);
  cast_qkvw_kernel<<<castGrid, blk, 0, stream>>>(swq, Wq_s);
  cast_qkvw_kernel<<<castGrid, blk, 0, stream>>>(swk, Wk_s);
  cast_qkvw_kernel<<<castGrid, blk, 0, stream>>>(swv, Wv_s);
  cast_qkvw_kernel<<<castGrid, blk, 0, stream>>>(cwq, Wq_c);
  cast_qkvw_kernel<<<castGrid, blk, 0, stream>>>(cwk, Wk_c);
  cast_qkvw_kernel<<<castGrid, blk, 0, stream>>>(cwv, Wv_c);
  cast_flat_kernel<<<castGrid, blk, 0, stream>>>(swo, Wo_s, (size_t)LNUM * MDIM * MDIM);
  cast_flat_kernel<<<castGrid, blk, 0, stream>>>(cwo, Wo_c, (size_t)LNUM * MDIM * MDIM);
  cast_flat_kernel<<<castGrid, blk, 0, stream>>>(w1, W1b, (size_t)LNUM * MDIM * FDIM);
  cast_flat_kernel<<<castGrid, blk, 0, stream>>>(w2, W2b, (size_t)LNUM * FDIM * MDIM);
  cast_flat_kernel<<<castGrid, blk, 0, stream>>>(enc, encB, (size_t)BDIM * TDIM * MDIM);

  hipMemcpyAsync(h, x, actF, hipMemcpyDeviceToDevice, stream);

  const int ROWS = BDIM * QDIM;                 // 4096
  const dim3 lnGrid(ROWS);
  const dim3 gemmM(MDIM / 128, ROWS / 128);     // (8,32)
  const dim3 gemmF(FDIM / 128, ROWS / 128);     // (32,32)
  const dim3 attGrid(QDIM / 128, BDIM * HNUM);  // (8,32)

  for (int l = 0; l < LNUM; ++l) {
    const size_t wA = (size_t)l * MDIM * (HNUM * DHEAD);
    const size_t wF1 = (size_t)l * MDIM * FDIM;
    const size_t wF2 = (size_t)l * FDIM * MDIM;

    // --- self attention ---
    ln_kernel<<<lnGrid, blk, 0, stream>>>(h, ln1g + l * MDIM, ln1b + l * MDIM, nB);
    gemm_bf16_kernel<<<gemmM, blk, 0, stream>>>(nB, Wq_s + wA, MDIM, MDIM,
        nullptr, nullptr, nullptr, nullptr, qB, 0);
    gemm_bf16_kernel<<<gemmM, blk, 0, stream>>>(nB, Wk_s + wA, MDIM, MDIM,
        nullptr, nullptr, nullptr, nullptr, kB, 0);
    gemm_bf16_kernel<<<gemmM, blk, 0, stream>>>(nB, Wv_s + wA, MDIM, MDIM,
        nullptr, nullptr, nullptr, nullptr, vB, 0);
    attn_kernel<<<attGrid, blk, 0, stream>>>(qB, kB, vB, preB, QDIM, 1);
    gemm_bf16_kernel<<<gemmM, blk, 0, stream>>>(preB, Wo_s + wA, MDIM, MDIM,
        nullptr, qmask, h, h1, nullptr, 0);

    // --- cross attention ---
    ln_kernel<<<lnGrid, blk, 0, stream>>>(h1, ln2g + l * MDIM, ln2b + l * MDIM, nB);
    gemm_bf16_kernel<<<gemmM, blk, 0, stream>>>(nB, Wq_c + wA, MDIM, MDIM,
        nullptr, nullptr, nullptr, nullptr, qB, 0);
    gemm_bf16_kernel<<<gemmM, blk, 0, stream>>>(encB, Wk_c + wA, MDIM, MDIM,
        nullptr, nullptr, nullptr, nullptr, kB, 0);
    gemm_bf16_kernel<<<gemmM, blk, 0, stream>>>(encB, Wv_c + wA, MDIM, MDIM,
        nullptr, nullptr, nullptr, nullptr, vB, 0);
    attn_kernel<<<attGrid, blk, 0, stream>>>(qB, kB, vB, preB, TDIM, 0);
    gemm_bf16_kernel<<<gemmM, blk, 0, stream>>>(preB, Wo_c + wA, MDIM, MDIM,
        nullptr, qmask, h1, h2, nullptr, 0);

    // --- FFN ---
    ln_kernel<<<lnGrid, blk, 0, stream>>>(h2, ln3g + l * MDIM, ln3b + l * MDIM, nB);
    gemm_bf16_kernel<<<gemmF, blk, 0, stream>>>(nB, W1b + wF1, MDIM, FDIM,
        b1 + (size_t)l * FDIM, nullptr, nullptr, nullptr, sB, 1);
    float* outPtr = (l == LNUM - 1) ? (float*)d_out : h;
    gemm_bf16_kernel<<<gemmM, blk, 0, stream>>>(sB, W2b + wF2, FDIM, MDIM,
        b2 + (size_t)l * MDIM, qmask, h2, outPtr, nullptr, 0);
  }
}
